// VectorQuantizer_11278584119645
// MI455X (gfx1250) — matrix operand order repaired and checked by reading, not compiled, not run on hardware
//
#include <hip/hip_runtime.h>
#include <hip/hip_bf16.h>
#include <stdint.h>

typedef __attribute__((ext_vector_type(16))) __bf16 v16bf;
typedef __attribute__((ext_vector_type(8)))  float  v8f;

#define TOKENS   16384
#define CDIM     256
#define KCODES   8192
#define ROWS_PB  128
#define KTILES   (KCODES / 16)
#define LDS_ROWQ 33          // uint4 per LDS B-row: 32 data + 1 pad (bank shift)
#define NSTAGE   4           // async pipeline depth (LDS buffers)

union FragAB {
    v16bf v;
    uint4 q[2];
};

static __device__ __forceinline__ unsigned int f2bf(float f) {
    unsigned int u = __float_as_uint(f);
    return (u + 0x7FFFu + ((u >> 16) & 1u)) >> 16;   // round-to-nearest-even
}

static __device__ __forceinline__ uint4 pack8bf(float4 a, float4 b) {
    uint4 o;
    o.x = f2bf(a.x) | (f2bf(a.y) << 16);
    o.y = f2bf(a.z) | (f2bf(a.w) << 16);
    o.z = f2bf(b.x) | (f2bf(b.y) << 16);
    o.w = f2bf(b.z) | (f2bf(b.w) << 16);
    return o;
}

// async DMA: 32 contiguous bytes global -> LDS (ASYNCcnt-tracked, no VGPR staging)
static __device__ __forceinline__ void async_copy32(unsigned ldsaddr, const void* gptr) {
    asm volatile(
        "global_load_async_to_lds_b128 %0, %1, off\n\t"
        "global_load_async_to_lds_b128 %0, %1, off offset:16"
        :: "v"(ldsaddr), "v"(gptr) : "memory");
}

// ---------------- prep 1: z (fp32) -> bf16 ----------------
__global__ __launch_bounds__(256) void vq_cvt_z(const float* __restrict__ z,
                                                unsigned short* __restrict__ zbf) {
    int t = blockIdx.x * 256 + threadIdx.x;          // each thread: 8 elements
    const float4* src = (const float4*)z;
    float4 a = src[t * 2];
    float4 b = src[t * 2 + 1];
    ((uint4*)zbf)[t] = pack8bf(a, b);
}

// ------- prep 2: embedding -> bf16 + per-code squared norm -------
__global__ __launch_bounds__(256) void vq_cvt_e(const float* __restrict__ e,
                                                unsigned short* __restrict__ ebf,
                                                float* __restrict__ enorm) {
    const int lane = threadIdx.x & 31;
    const int wave = threadIdx.x >> 5;
    const int row  = blockIdx.x * 8 + wave;          // one wave per code row
    const float4* src = (const float4*)(e + (size_t)row * CDIM);
    float4 a = src[lane * 2];
    float4 b = src[lane * 2 + 1];
    float s = a.x * a.x + a.y * a.y + a.z * a.z + a.w * a.w
            + b.x * b.x + b.y * b.y + b.z * b.z + b.w * b.w;
    ((uint4*)(ebf + (size_t)row * CDIM))[lane] = pack8bf(a, b);
    #pragma unroll
    for (int off = 16; off > 0; off >>= 1) s += __shfl_xor(s, off, 32);
    if (lane == 0) enorm[row] = s;
}

// ---------------- main: WMMA distance + argmin + gather ----------------
__global__ __launch_bounds__(256) void vq_argmin_gather(
    const unsigned short* __restrict__ zbf,
    const unsigned short* __restrict__ ebf,
    const float* __restrict__ enorm,
    const float* __restrict__ emb_f32,
    float* __restrict__ out)
{
    __shared__ uint4 ldsB[NSTAGE][16 * LDS_ROWQ];   // 4 x (16 codes x 512B, padded)

    const int tid  = threadIdx.x;
    const int lane = tid & 31;
    const int wave = tid >> 5;
    const int half = lane >> 4;     // 0: lanes 0-15, 1: lanes 16-31
    const int l15  = lane & 15;

    const int rbase = blockIdx.x * ROWS_PB + wave * 16;

    // ---- A fragments: 16 rows x 256 cols bf16, contiguous k = 16*half + s, same order as B ----
    FragAB afrag[8];
    {
        const uint4* zrow = (const uint4*)(zbf + (size_t)(rbase + l15) * CDIM);
        #pragma unroll
        for (int q = 0; q < 8; ++q) {
            afrag[q].q[0] = zrow[q * 4 + half * 2];       // K = half*16 .. +7
            afrag[q].q[1] = zrow[q * 4 + half * 2 + 1];   // K = half*16+8 .. +15
        }
    }

    float minv[8];
    int   mini[8];
    #pragma unroll
    for (int i = 0; i < 8; ++i) { minv[i] = 3.4e38f; mini[i] = 0; }

    // cooperative async tile copy: 256 threads x 32B = 16 codes x 512B
    const int csub = tid >> 4;              // code within tile
    const unsigned ldsbase  = (unsigned)(size_t)&ldsB[0][0];   // low 32 bits = LDS offset
    const unsigned ldsstage = (unsigned)(16 * LDS_ROWQ * sizeof(uint4));
    const unsigned ldschunk = (unsigned)((csub * LDS_ROWQ + (tid & 15) * 2) * sizeof(uint4));

    // prologue: launch first 2 tiles
    #pragma unroll
    for (int t = 0; t < 2; ++t) {
        const char* g = (const char*)(ebf + (size_t)(t * 16 + csub) * CDIM) + (tid & 15) * 32;
        async_copy32(ldsbase + (unsigned)t * ldsstage + ldschunk, g);
    }

    const int NPAIR = KTILES / 2;   // 2 tiles per barrier window
    for (int p = 0; p < NPAIR; ++p) {
        const int kt0 = 2 * p;
        if (p + 1 < NPAIR) {        // issue the next two tiles, 4 deep in flight
            #pragma unroll
            for (int u = 2; u < 4; ++u) {
                const int nt = kt0 + u;
                const char* g = (const char*)(ebf + (size_t)(nt * 16 + csub) * CDIM)
                              + (tid & 15) * 32;
                async_copy32(ldsbase + (unsigned)(nt & (NSTAGE - 1)) * ldsstage + ldschunk, g);
            }
            // asyncs retire in order: cnt<=4 => the two oldest tiles (kt0, kt0+1) landed
            asm volatile("s_wait_asynccnt 0x4" ::: "memory");
        } else {
            asm volatile("s_wait_asynccnt 0x0" ::: "memory");
        }
        __syncthreads();   // all waves' chunks of tiles kt0/kt0+1 visible

        #pragma unroll
        for (int u = 0; u < 2; ++u) {
            const int kt = kt0 + u;
            const uint4* buf = &ldsB[kt & (NSTAGE - 1)][0];
            const int bbase = l15 * LDS_ROWQ + half * 2;

            FragAB b[2];           // double-buffered B fragments: LDS hides under WMMA
            b[0].q[0] = buf[bbase];
            b[0].q[1] = buf[bbase + 1];
            v8f acc = {0.f, 0.f, 0.f, 0.f, 0.f, 0.f, 0.f, 0.f};
            #pragma unroll
            for (int q = 0; q < 8; ++q) {
                if (q + 1 < 8) {
                    b[(q + 1) & 1].q[0] = buf[bbase + (q + 1) * 4];
                    b[(q + 1) & 1].q[1] = buf[bbase + (q + 1) * 4 + 1];
                }
                acc = __builtin_amdgcn_wmma_f32_16x16x32_bf16(
                          false, afrag[q].v, false, b[q & 1].v, (short)0, acc, false, false);
            }

            const float en   = enorm[kt * 16 + l15];   // per-lane, one cacheline
            const int   code = kt * 16 + l15;
            #pragma unroll
            for (int i = 0; i < 8; ++i) {
                float d = fmaf(-2.0f, acc[i], en);     // ||e||^2 - 2 z.e
                if (d < minv[i]) { minv[i] = d; mini[i] = code; }
            }
        }
        __syncthreads();   // all waves done reading before these buffers are rewritten
    }

    // ---- argmin across the 16 lanes holding each row (first-index ties) ----
    #pragma unroll
    for (int i = 0; i < 8; ++i) {
        #pragma unroll
        for (int off = 1; off < 16; off <<= 1) {
            float od = __shfl_xor(minv[i], off, 32);
            int   oi = __shfl_xor(mini[i], off, 32);
            if (od < minv[i] || (od == minv[i] && oi < mini[i])) {
                minv[i] = od; mini[i] = oi;
            }
        }
    }

    int rowcode[16];
    #pragma unroll
    for (int i = 0; i < 8; ++i) {
        rowcode[i]     = __shfl(mini[i], 0, 32);    // rows 0-7  live in lanes 0-15
        rowcode[i + 8] = __shfl(mini[i], 16, 32);   // rows 8-15 live in lanes 16-31
    }

    // ---- gather fp32 embedding rows -> output ----
    #pragma unroll 4
    for (int m = 0; m < 16; ++m) {
        const float4* src = (const float4*)(emb_f32 + (size_t)rowcode[m] * CDIM);
        float4*       dst = (float4*)(out + (size_t)(rbase + m) * CDIM);
        dst[lane]      = src[lane];
        dst[lane + 32] = src[lane + 32];
    }
}

extern "C" void kernel_launch(void* const* d_in, const int* in_sizes, int n_in,
                              void* d_out, int out_size, void* d_ws, size_t ws_size,
                              hipStream_t stream) {
    const float* z   = (const float*)d_in[0];   // [1024,16,256] fp32
    const float* emb = (const float*)d_in[1];   // [8192,256]    fp32
    float* out = (float*)d_out;

    // workspace layout: zbf16 (8 MiB) | ebf16 (4 MiB) | enorm (32 KiB)
    unsigned short* zbf   = (unsigned short*)d_ws;
    unsigned short* ebf   = zbf + (size_t)TOKENS * CDIM;
    float*          enorm = (float*)(ebf + (size_t)KCODES * CDIM);

    vq_cvt_z<<<(TOKENS * CDIM) / 8 / 256, 256, 0, stream>>>(z, zbf);
    vq_cvt_e<<<KCODES / 8, 256, 0, stream>>>(emb, ebf, enorm);
    vq_argmin_gather<<<TOKENS / ROWS_PB, 256, 0, stream>>>(zbf, ebf, enorm, emb, out);
}
